// OriginalScorer_11287174054653
// MI455X (gfx1250) — matrix operand order repaired and checked by reading, not compiled, not run on hardware
//
#include <hip/hip_runtime.h>
#include <hip/hip_bf16.h>
#include <stdint.h>

typedef __bf16 bf16_t;
typedef bf16_t bf16x16 __attribute__((ext_vector_type(16)));
typedef float  f32x8   __attribute__((ext_vector_type(8)));
typedef unsigned short u16;
typedef unsigned int   u32;

#define CDIM 512     // feature dim (K)
#define HW   1024    // H*W per image
#define BROW2 1048   // u16 per LDS row: 512 hi + 512 lo + 24 pad (stride=524 dw === 12 mod 64 banks)
#define TOPK 9

static __device__ __forceinline__ u16 f32_to_bf16_rne(float x) {
  u32 u = __float_as_uint(x);
  return (u16)((u + 0x7FFFu + ((u >> 16) & 1u)) >> 16);
}

// Async global->LDS copy of 16B per lane (CDNA5 ASYNCcnt path).
static __device__ __forceinline__ void async_ld16(const u16* gsrc, unsigned ldsOff) {
  asm volatile("global_load_async_to_lds_b128 %0, %1, off"
               :: "v"(ldsOff), "v"(gsrc)
               : "memory");
}
static __device__ __forceinline__ void wait_async0() {
  asm volatile("s_wait_asynccnt 0x0" ::: "memory");
}

// ---------------------------------------------------------------------------
// 1) Split fp32 -> bf16 hi/lo pair (error-compensated representation).
// ---------------------------------------------------------------------------
__global__ void k_cvt_split(const float* __restrict__ src, int srcCount, int totCount,
                            u16* __restrict__ hi, u16* __restrict__ lo) {
  int i = blockIdx.x * blockDim.x + threadIdx.x;
  if (i >= totCount) return;
  float x = (i < srcCount) ? src[i] : 0.f;
  u16 h = f32_to_bf16_rne(x);
  float fh = __uint_as_float((u32)h << 16);
  hi[i] = h;
  lo[i] = f32_to_bf16_rne(x - fh);
}

// ---------------------------------------------------------------------------
// 2) Exact fp32 row squared-norms; one wave32 per row. Pad rows get padVal.
// ---------------------------------------------------------------------------
__global__ void k_row_norm(const float* __restrict__ X, float* __restrict__ out,
                           int nrows, int padRows, float padVal) {
  int gwave = (blockIdx.x * blockDim.x + threadIdx.x) >> 5;
  int lane  = threadIdx.x & 31;
  if (gwave >= padRows) return;
  if (gwave >= nrows) { if (lane == 0) out[gwave] = padVal; return; }
  const float* row = X + (size_t)gwave * CDIM;
  float s = 0.f;
  for (int k = lane; k < CDIM; k += 32) { float v = row[k]; s = fmaf(v, v, s); }
  #pragma unroll
  for (int off = 16; off > 0; off >>= 1) s += __shfl_xor(s, off, 32);
  if (lane == 0) out[gwave] = s;
}

// ---------------------------------------------------------------------------
// 3) Main compute: min over all mb rows of ||a-b||^2 via bf16 hi/lo WMMA.
//    Block = 256 threads = 8 wave32; each wave owns a 16-row M tile.
//    mb tiles (16 rows, hi+lo) are double-buffered in LDS and streamed with
//    GLOBAL_LOAD_ASYNC_TO_LDS_B128 so the DMA for tile nt+1 overlaps the
//    v_wmma_f32_16x16x32_bf16 work on tile nt. One barrier per tile.
// ---------------------------------------------------------------------------
__global__ __launch_bounds__(256)
void k_min_dist(const u16* __restrict__ fvHi, const u16* __restrict__ fvLo,
                const u16* __restrict__ mbHi, const u16* __restrict__ mbLo,
                const float* __restrict__ fvNorm, const float* __restrict__ mbNorm,
                float* __restrict__ pixOut, int nChunks) {
  // [buf][mb-row-in-tile][0..512)=hi, [512..1024)=lo, 24 u16 pad
  __shared__ u16 sB[2][16][BROW2];

  const int tid  = threadIdx.x;
  const int lane = tid & 31;
  const int wave = tid >> 5;
  const int lr   = lane & 15;   // row-in-tile (A) / col-in-tile (B/C)
  const int lh   = lane >> 4;   // half-wave selector
  const int mtile = blockIdx.x * 128 + wave * 16;

  float fvn[8];
  #pragma unroll
  for (int r = 0; r < 8; ++r) fvn[r] = fvNorm[mtile + r + 8 * lh];

  float minv[8];
  #pragma unroll
  for (int r = 0; r < 8; ++r) minv[r] = 3.4e38f;

  const u16* aHiRow = fvHi + (size_t)(mtile + lr) * CDIM;
  const u16* aLoRow = fvLo + (size_t)(mtile + lr) * CDIM;

  // issue async stage of one 16-row mb tile (hi+lo) into LDS buffer `buf`
  auto stage = [&](int nt, int buf) {
    const size_t srcBase = (size_t)nt * 16 * CDIM;
    #pragma unroll
    for (int j = 0; j < 4; ++j) {
      int i    = tid + j * 256;        // 16B granule id, 0..1023
      int half = i >> 9;               // 0 = hi, 1 = lo
      int idx  = i & 511;
      int row  = idx >> 5;             // 32 granules (256 u16) per row
      int c8   = (idx & 31) * 8;       // u16 offset within row
      const u16* src = (half ? mbLo : mbHi) + srcBase + (size_t)row * CDIM + c8;
      unsigned dst = (unsigned)(uintptr_t)&sB[buf][row][half * CDIM + c8];
      async_ld16(src, dst);
    }
  };

  stage(0, 0);

  for (int nt = 0; nt < nChunks; ++nt) {
    const int buf = nt & 1;
    wait_async0();        // this wave's loads for tile nt have landed
    __syncthreads();      // everyone's landed; everyone done reading buf^1
    if (nt + 1 < nChunks) stage(nt + 1, buf ^ 1);   // overlap DMA with WMMAs

    f32x8 acc = {0.f, 0.f, 0.f, 0.f, 0.f, 0.f, 0.f, 0.f};
    #pragma unroll 4
    for (int ks = 0; ks < CDIM; ks += 32) {
      // A fragment (16-bit A layout): lanes 0-15 hold K [0,8)+[16,24),
      // lanes 16-31 hold K [8,16)+[24,32) of row (mtile+lr).
      union { uint4 q[2]; bf16x16 v; } aH, aL, bH, bL;
      const u16* pa = aHiRow + ks + lh * 8;
      aH.q[0] = *(const uint4*)pa;        aH.q[1] = *(const uint4*)(pa + 16);
      const u16* pl = aLoRow + ks + lh * 8;
      aL.q[0] = *(const uint4*)pl;        aL.q[1] = *(const uint4*)(pl + 16);
      // B fragment (16-bit B layout): lanes 0-15 hold K [0,8)+[16,24),
      // lanes 16-31 hold K [8,16)+[24,32) of column (mb row) lr.
      const u16* qh = &sB[buf][lr][ks + lh * 8];
      bH.q[0] = *(const uint4*)qh;        bH.q[1] = *(const uint4*)(qh + 16);
      const u16* ql = &sB[buf][lr][CDIM + ks + lh * 8];
      bL.q[0] = *(const uint4*)ql;        bL.q[1] = *(const uint4*)(ql + 16);

      acc = __builtin_amdgcn_wmma_f32_16x16x32_bf16(false, aH.v, false, bH.v, (short)0, acc, false, false);
      acc = __builtin_amdgcn_wmma_f32_16x16x32_bf16(false, aH.v, false, bL.v, (short)0, acc, false, false);
      acc = __builtin_amdgcn_wmma_f32_16x16x32_bf16(false, aL.v, false, bH.v, (short)0, acc, false, false);
    }

    // sq_dist(m,n) = ||a_m||^2 + ||b_n||^2 - 2*dot ; running min over n
    float mbn = mbNorm[nt * 16 + lr];
    #pragma unroll
    for (int r = 0; r < 8; ++r) {
      float d = fvn[r] + mbn - 2.f * acc[r];
      minv[r] = fminf(minv[r], d);
    }
  }

  // reduce min across the 16 lanes (each holds a different n column)
  #pragma unroll
  for (int r = 0; r < 8; ++r) {
    float v = minv[r];
    #pragma unroll
    for (int off = 8; off > 0; off >>= 1) v = fminf(v, __shfl_xor(v, off, 16));
    minv[r] = v;
  }
  if (lr == 0) {
    #pragma unroll
    for (int r = 0; r < 8; ++r)
      pixOut[mtile + r + 8 * lh] = sqrtf(fmaxf(minv[r], 0.f));
  }
}

// ---------------------------------------------------------------------------
// 4) Per-image argmax over HW pixel scores (first-index tie break).
// ---------------------------------------------------------------------------
__global__ void k_argmax(const float* __restrict__ pix, int* __restrict__ flatIdx, int hw) {
  __shared__ float sv[256];
  __shared__ int   si[256];
  int img = blockIdx.x, tid = threadIdx.x;
  const float* p = pix + (size_t)img * hw;
  int per  = hw / blockDim.x;       // contiguous chunk per thread
  int base = tid * per;
  float bv = -3.4e38f; int bi = base;
  for (int j = 0; j < per; ++j) {
    float v = p[base + j];
    if (v > bv) { bv = v; bi = base + j; }
  }
  sv[tid] = bv; si[tid] = bi;
  __syncthreads();
  if (tid == 0) {
    float best = sv[0]; int bidx = si[0];
    for (int t = 1; t < (int)blockDim.x; ++t)
      if (sv[t] > best) { best = sv[t]; bidx = si[t]; }
    flatIdx[img] = img * hw + bidx;
  }
}

// ---------------------------------------------------------------------------
// 5) Exact fp32 squared distances from each selected row to all mb rows.
// ---------------------------------------------------------------------------
__global__ void k_sel_dists(const float* __restrict__ fv, const float* __restrict__ mb,
                            const int* __restrict__ flatIdx, float* __restrict__ dists, int N) {
  int img = blockIdx.y;
  int n = blockIdx.x * blockDim.x + threadIdx.x;
  if (n >= N) return;
  const float4* s = (const float4*)(fv + (size_t)flatIdx[img] * CDIM);
  const float4* m = (const float4*)(mb + (size_t)n * CDIM);
  float acc = 0.f;
  #pragma unroll 4
  for (int k = 0; k < CDIM / 4; ++k) {
    float4 a = s[k], b4 = m[k];
    float dx = a.x - b4.x, dy = a.y - b4.y, dz = a.z - b4.z, dw = a.w - b4.w;
    acc = fmaf(dx, dx, acc); acc = fmaf(dy, dy, acc);
    acc = fmaf(dz, dz, acc); acc = fmaf(dw, dw, acc);
  }
  dists[(size_t)img * N + n] = acc;
}

// ---------------------------------------------------------------------------
// 6) Per-image top-9 smallest (ascending) + softmax-weighted score.
// ---------------------------------------------------------------------------
__global__ void k_topk_score(const float* __restrict__ dists, const int* __restrict__ bPtr,
                             float* __restrict__ imgOut, int N) {
  __shared__ float sTop[256 * TOPK];
  int img = blockIdx.x, tid = threadIdx.x;
  const float* d = dists + (size_t)img * N;
  float top[TOPK];
  #pragma unroll
  for (int i = 0; i < TOPK; ++i) top[i] = 3.4e38f;
  for (int n = tid; n < N; n += blockDim.x) {
    float v = d[n];
    if (v < top[TOPK - 1]) {
      int j = TOPK - 1;
      while (j > 0 && top[j - 1] > v) { top[j] = top[j - 1]; --j; }
      top[j] = v;
    }
  }
  #pragma unroll
  for (int i = 0; i < TOPK; ++i) sTop[tid * TOPK + i] = top[i];
  __syncthreads();
  if (tid == 0) {
    int b = *bPtr; if (b > TOPK) b = TOPK; if (b < 1) b = 1;
    int total = blockDim.x * TOPK;
    float sel[TOPK];
    for (int i = 0; i < b; ++i) {            // repeated extract-min
      float mv = 3.4e38f; int mj = 0;
      for (int j = 0; j < total; ++j)
        if (sTop[j] < mv) { mv = sTop[j]; mj = j; }
      sel[i] = sqrtf(fmaxf(mv, 0.f));
      sTop[mj] = 3.4e38f;
    }
    float score = sel[0];
    if (b > 1) {
      float mx = sel[0];
      for (int i = 1; i < b; ++i) mx = fmaxf(mx, sel[i]);
      float sum = 0.f;
      for (int i = 0; i < b; ++i) sum += expf(sel[i] - mx);
      float p0 = expf(sel[0] - mx) / sum;
      score = score * (1.f - p0);
    }
    imgOut[img] = score;
  }
}

// ---------------------------------------------------------------------------
extern "C" void kernel_launch(void* const* d_in, const int* in_sizes, int n_in,
                              void* d_out, int out_size, void* d_ws, size_t ws_size,
                              hipStream_t stream) {
  const float* fv   = (const float*)d_in[0];   // (B*H*W, 512) flat
  const float* mb   = (const float*)d_in[1];   // (N, 512) flat
  const int*   bPtr = (const int*)d_in[2];

  const int M    = in_sizes[0] / CDIM;         // 8192
  const int N    = in_sizes[1] / CDIM;         // 25000
  const int Npad = (N + 15) & ~15;             // 25008
  const int B    = M / HW;                     // 8

  float* outPix = (float*)d_out;               // M pixel scores (B,1,H,W flat)
  float* outImg = (float*)d_out + M;           // B image scores

  // workspace carve-up (~69 MB)
  char* ws = (char*)d_ws;
  size_t off = 0;
  auto carve = [&](size_t bytes) -> void* {
    void* p = ws + off;
    off = (off + bytes + 255) & ~(size_t)255;
    return p;
  };
  u16*   mbHi    = (u16*)carve((size_t)Npad * CDIM * sizeof(u16));
  u16*   mbLo    = (u16*)carve((size_t)Npad * CDIM * sizeof(u16));
  u16*   fvHi    = (u16*)carve((size_t)M * CDIM * sizeof(u16));
  u16*   fvLo    = (u16*)carve((size_t)M * CDIM * sizeof(u16));
  float* mbNorm  = (float*)carve((size_t)Npad * sizeof(float));
  float* fvNorm  = (float*)carve((size_t)M * sizeof(float));
  int*   flatIdx = (int*)carve((size_t)B * sizeof(int));
  float* dists   = (float*)carve((size_t)B * N * sizeof(float));
  (void)ws_size; (void)n_in; (void)out_size;

  // 1) bf16 hi/lo splits
  {
    int tot = M * CDIM;
    k_cvt_split<<<(tot + 255) / 256, 256, 0, stream>>>(fv, tot, tot, fvHi, fvLo);
  }
  {
    int tot = Npad * CDIM, srcN = N * CDIM;
    k_cvt_split<<<(tot + 255) / 256, 256, 0, stream>>>(mb, srcN, tot, mbHi, mbLo);
  }
  // 2) exact fp32 norms (pad mb rows get 1e30 so they never win the min)
  k_row_norm<<<(M * 32 + 255) / 256, 256, 0, stream>>>(fv, fvNorm, M, M, 0.f);
  k_row_norm<<<(Npad * 32 + 255) / 256, 256, 0, stream>>>(mb, mbNorm, N, Npad, 1e30f);
  // 3) WMMA min-distance sweep -> pixel scores
  k_min_dist<<<M / 128, 256, 0, stream>>>(fvHi, fvLo, mbHi, mbLo,
                                          fvNorm, mbNorm, outPix, Npad / 16);
  // 4) per-image argmax
  k_argmax<<<B, 256, 0, stream>>>(outPix, flatIdx, HW);
  // 5) exact distances for selected rows
  dim3 g5((N + 255) / 256, B);
  k_sel_dists<<<g5, 256, 0, stream>>>(fv, mb, flatIdx, dists, N);
  // 6) top-k + softmax score
  k_topk_score<<<B, 256, 0, stream>>>(dists, bPtr, outImg, N);
}
